// DeepseekV4DecoderLayer_14886356648850
// MI455X (gfx1250) — matrix operand order repaired and checked by reading, not compiled, not run on hardware
//
#include <hip/hip_runtime.h>

#define T_    8192
#define HC_   4
#define H_    1024
#define EENG_ 512
#define DFF_  2048
#define NHEADS_ 8
#define NPER_ 131072
#define EDIM_ 64

typedef __bf16 bf16_t;
typedef __attribute__((ext_vector_type(16))) __bf16 v16bf;
typedef __attribute__((ext_vector_type(8)))  __bf16 v8bf;
typedef __attribute__((ext_vector_type(8)))  float  v8f;

static __device__ __forceinline__ bf16_t f2bf(float f) {
    union { float f; unsigned u; } v; v.f = f;
    unsigned r = v.u + 0x7fffu + ((v.u >> 16) & 1u);   // round-to-nearest-even
    unsigned short s = (unsigned short)(r >> 16);
    bf16_t out;
    __builtin_memcpy(&out, &s, 2);
    return out;
}

static __device__ __forceinline__ float sigmoidf_(float x) { return 1.0f / (1.0f + __expf(-x)); }
static __device__ __forceinline__ float siluf_(float x)    { return x / (1.0f + __expf(-x)); }

// ---------------------------------------------------------------------------
// f32 [K,N] -> bf16 [N,K] transpose-convert (weight prep, one-shot)
// ---------------------------------------------------------------------------
__global__ void transpose_bf16(const float* __restrict__ src, bf16_t* __restrict__ dst,
                               int Kd, int N) {
    long i = (long)blockIdx.x * 256 + threadIdx.x;
    long tot = (long)Kd * N;
    if (i >= tot) return;
    long n = i / Kd, k = i - n * Kd;
    dst[i] = f2bf(src[k * (long)N + n]);
}

// ---------------------------------------------------------------------------
// Hash-embedding gather -> bf16 activation matrix [T, 512]
// ---------------------------------------------------------------------------
__global__ void gather_emb(const int* __restrict__ ids, const float* __restrict__ table,
                           bf16_t* __restrict__ emb) {
    int t = blockIdx.x;
    for (int e = threadIdx.x; e < EENG_; e += 256) {
        int hh = e >> 6;          // head = e / 64
        int c  = e & 63;
        long row = (long)ids[t * NHEADS_ + hh] + (long)hh * NPER_;
        emb[(long)t * EENG_ + e] = f2bf(table[row * EDIM_ + c]);
    }
}

// ---------------------------------------------------------------------------
// WMMA GEMM: C[M,N] f32 = A[M,K] bf16 (row-major) x Bt[N,K] bf16 (B transposed)
// One wave computes a 16x64 strip: 1 A fragment feeds 4 v_wmma_f32_16x16x32_bf16.
// Software-pipelined (double-buffered fragments) so the K+32 loads are in
// flight while the K-step WMMAs execute.
// M%16==0, N%64==0, K%64==0. blockDim=256 (8 waves). blockIdx.y = batch.
// ---------------------------------------------------------------------------
static __device__ __forceinline__ v16bf load_a_frag(const bf16_t* __restrict__ arow, int k0) {
    v8bf lo = *(const v8bf*)(arow + k0);
    v8bf hi = *(const v8bf*)(arow + k0 + 8);
    v16bf af;
#pragma unroll
    for (int i = 0; i < 8; ++i) { af[i] = lo[i]; af[i + 8] = hi[i]; }
    return af;
}

__global__ void gemm_bf16_wmma(const bf16_t* __restrict__ A, const bf16_t* __restrict__ Bt,
                               float* __restrict__ C, int M, int N, int Kd,
                               long long sB, long long sC) {
    const bf16_t* Bb = Bt + (long)blockIdx.y * (long)sB;
    float*        Cb = C  + (long)blockIdx.y * (long)sC;

    const int lane = threadIdx.x & 31;
    const int wave = threadIdx.x >> 5;
    const int tiles_n = N >> 6;
    const int tile = blockIdx.x * (blockDim.x >> 5) + wave;
    const int tm = tile / tiles_n;
    const int tn = tile % tiles_n;
    if (tm >= (M >> 4)) return;

    const int half = lane >> 4;       // 0: lanes 0-15, 1: lanes 16-31
    const int l    = lane & 15;

    // A fragment: row M = tm*16+l; half 0 holds K=0..15, half 1 K=16..31
    const bf16_t* arow = A + (long)(tm * 16 + l) * Kd + half * 16;
    // B fragment (Bt row = column of B): half 0 holds K=0..15, half 1 K=16..31
    const bf16_t* brow = Bb + (long)(tn * 64 + l) * Kd + half * 16;
    const long bs = (long)16 * Kd;    // Bt row-group stride (16 columns of B)

    v8f acc0 = {}, acc1 = {}, acc2 = {}, acc3 = {};

    // prologue: fragments for K-step 0
    v16bf aC  = load_a_frag(arow, 0);
    v16bf bC0 = *(const v16bf*)(brow);
    v16bf bC1 = *(const v16bf*)(brow + bs);
    v16bf bC2 = *(const v16bf*)(brow + 2 * bs);
    v16bf bC3 = *(const v16bf*)(brow + 3 * bs);

    int k0 = 0;
    for (; k0 < Kd - 32; k0 += 32) {
        const int kn = k0 + 32;
        // issue next-step loads before consuming current fragments
        v16bf aN  = load_a_frag(arow, kn);
        v16bf bN0 = *(const v16bf*)(brow + kn);
        v16bf bN1 = *(const v16bf*)(brow + bs + kn);
        v16bf bN2 = *(const v16bf*)(brow + 2 * bs + kn);
        v16bf bN3 = *(const v16bf*)(brow + 3 * bs + kn);
        __builtin_prefetch(arow + kn + 32, 0, 3);
        __builtin_prefetch(brow + kn + 32, 0, 3);

        acc0 = __builtin_amdgcn_wmma_f32_16x16x32_bf16(false, aC, false, bC0, (short)0, acc0, false, false);
        acc1 = __builtin_amdgcn_wmma_f32_16x16x32_bf16(false, aC, false, bC1, (short)0, acc1, false, false);
        acc2 = __builtin_amdgcn_wmma_f32_16x16x32_bf16(false, aC, false, bC2, (short)0, acc2, false, false);
        acc3 = __builtin_amdgcn_wmma_f32_16x16x32_bf16(false, aC, false, bC3, (short)0, acc3, false, false);

        aC = aN; bC0 = bN0; bC1 = bN1; bC2 = bN2; bC3 = bN3;
    }
    // epilogue: last K-step
    acc0 = __builtin_amdgcn_wmma_f32_16x16x32_bf16(false, aC, false, bC0, (short)0, acc0, false, false);
    acc1 = __builtin_amdgcn_wmma_f32_16x16x32_bf16(false, aC, false, bC1, (short)0, acc1, false, false);
    acc2 = __builtin_amdgcn_wmma_f32_16x16x32_bf16(false, aC, false, bC2, (short)0, acc2, false, false);
    acc3 = __builtin_amdgcn_wmma_f32_16x16x32_bf16(false, aC, false, bC3, (short)0, acc3, false, false);

    // C/D layout: lanes 0-15 -> rows tm*16+0..7, lanes 16-31 -> rows tm*16+8..15, col = l
    float* crow = Cb + (long)(tm * 16 + half * 8) * N + tn * 64 + l;
#pragma unroll
    for (int r = 0; r < 8; ++r) {
        crow[(long)r * N +  0] = acc0[r];
        crow[(long)r * N + 16] = acc1[r];
        crow[(long)r * N + 32] = acc2[r];
        crow[(long)r * N + 48] = acc3[r];
    }
}

// ---------------------------------------------------------------------------
// msq[t] = mean(P[t,:]^2) over H
// ---------------------------------------------------------------------------
__global__ void row_meansq(const float* __restrict__ P, float* __restrict__ msq) {
    int t = blockIdx.x, tid = threadIdx.x;
    __shared__ float red[256];
    const float* x = P + (long)t * H_;
    float s = 0.f;
#pragma unroll
    for (int q = 0; q < 4; ++q) { float v = x[tid + q * 256]; s += v * v; }
    red[tid] = s; __syncthreads();
    for (int st = 128; st > 0; st >>= 1) { if (tid < st) red[tid] += red[tid + st]; __syncthreads(); }
    if (tid == 0) msq[t] = red[0] * (1.0f / H_);
}

// ---------------------------------------------------------------------------
// Engram gate: gate[t,g] = sigmoid(dot(grms(keys), grms(query)) / 32)
// a[t,g] = gate * rsqrt(gate^2 * msq[t] + eps)   (conv-norm scale, value factored)
// ---------------------------------------------------------------------------
__global__ void gate_kernel(const float* __restrict__ keys, const float* __restrict__ hid,
                            const float* __restrict__ kg, const float* __restrict__ qg,
                            const float* __restrict__ msq,
                            float* __restrict__ gate, float* __restrict__ aout) {
    int t = blockIdx.x, g = blockIdx.y, tid = threadIdx.x;  // 128 threads
    __shared__ float r0[128], r1[128], r2[128];
    const float* kr  = keys + ((long)g * T_ + t) * H_;
    const float* qr  = hid  + ((long)t * HC_ + g) * H_;
    const float* kgg = kg + g * H_;
    const float* qgg = qg + g * H_;
    float sk = 0.f, sq = 0.f, sd = 0.f;
    for (int c = tid; c < H_; c += 128) {
        float kv = kr[c], qv = qr[c];
        sk += kv * kv; sq += qv * qv;
        sd += kv * kgg[c] * qv * qgg[c];
    }
    r0[tid] = sk; r1[tid] = sq; r2[tid] = sd; __syncthreads();
    for (int s = 64; s > 0; s >>= 1) {
        if (tid < s) { r0[tid] += r0[tid + s]; r1[tid] += r1[tid + s]; r2[tid] += r2[tid + s]; }
        __syncthreads();
    }
    if (tid == 0) {
        float rk = rsqrtf(r0[0] * (1.0f / H_) + 1e-6f);
        float rq = rsqrtf(r1[0] * (1.0f / H_) + 1e-6f);
        float gt = sigmoidf_(r2[0] * rk * rq * (1.0f / 32.0f));   // sqrt(1024)=32
        gate[(long)t * HC_ + g] = gt;
        aout[(long)t * HC_ + g] = gt * rsqrtf(gt * gt * msq[t] + 1e-6f);
    }
}

// ---------------------------------------------------------------------------
// h = hidden + value + silu(causal dilated depthwise conv of normalized value)
// value[t,g,c] = gate[t,g]*P[t,c];  v_norm tap = a[ts,g]*P[ts,c]*conv_g[g,c]
// ---------------------------------------------------------------------------
__global__ void build_h(const float* __restrict__ hid, const float* __restrict__ P,
                        const float* __restrict__ gate, const float* __restrict__ a,
                        const float* __restrict__ cw, const float* __restrict__ cg,
                        float* __restrict__ h) {
    long idx = (long)blockIdx.x * 256 + threadIdx.x;
    if (idx >= (long)T_ * HC_ * H_) return;
    int  c = idx & (H_ - 1);
    long tmp = idx >> 10;
    int  g = tmp & (HC_ - 1);
    long t = tmp >> 2;
    int  ch = g * H_ + c;
    const float* w = cw + (long)ch * 4;
    float acc = 0.f;
#pragma unroll
    for (int k = 0; k < 4; ++k) {
        long ts = t - (long)(3 - k) * 3;   // dilation = 3
        if (ts >= 0) acc += w[k] * (a[ts * HC_ + g] * P[ts * H_ + c]);
    }
    acc *= cg[ch];
    h[idx] = hid[idx] + gate[t * HC_ + g] * P[t * H_ + c] + siluf_(acc);
}

// ---------------------------------------------------------------------------
// MHC head per token: h_out = xf@W (4096x24), h_pre/h_post/h_res, 4x4 Sinkhorn,
// hp = sum_i h_pre[i]*h[t,i,:], pre-RMS -> bf16
// ---------------------------------------------------------------------------
__global__ void mhc_kernel(const float* __restrict__ h, const float* __restrict__ W,
                           const float* __restrict__ b, const float* __restrict__ alpha,
                           const float* __restrict__ preg,
                           float* __restrict__ hres, float* __restrict__ hpost,
                           bf16_t* __restrict__ hp_bf) {
    const int t = blockIdx.x, tid = threadIdx.x;  // 256 threads
    const float* x = h + (long)t * (HC_ * H_);
    __shared__ float red[256];
    __shared__ float s_out[24];
    __shared__ float s_ssq;
    __shared__ float s_pre[4];

    float acc[24];
#pragma unroll
    for (int o = 0; o < 24; ++o) acc[o] = 0.f;
    float ssq = 0.f;
    for (int e = tid; e < HC_ * H_; e += 256) {
        float xv = x[e]; ssq += xv * xv;
        const float* wr = W + (long)e * 24;
#pragma unroll
        for (int o = 0; o < 24; ++o) acc[o] += xv * wr[o];
    }
    red[tid] = ssq; __syncthreads();
    for (int s = 128; s > 0; s >>= 1) { if (tid < s) red[tid] += red[tid + s]; __syncthreads(); }
    if (tid == 0) s_ssq = red[0];
    __syncthreads();
    for (int o = 0; o < 24; ++o) {
        red[tid] = acc[o]; __syncthreads();
        for (int s = 128; s > 0; s >>= 1) { if (tid < s) red[tid] += red[tid + s]; __syncthreads(); }
        if (tid == 0) s_out[o] = red[0];
        __syncthreads();
    }

    if (tid == 0) {
        float rinv = rsqrtf(s_ssq * (1.0f / (HC_ * H_)));   // reference r_inv has no eps
        float a0 = alpha[0], a1 = alpha[1], a2 = alpha[2];
        float res[16], u[4] = {1,1,1,1}, v[4] = {1,1,1,1};
#pragma unroll
        for (int i = 0; i < 4; ++i) s_pre[i] = sigmoidf_(rinv * a0 * s_out[i] + b[i]);
#pragma unroll
        for (int i = 0; i < 4; ++i) hpost[(long)t * 4 + i] = 2.0f * sigmoidf_(rinv * a1 * s_out[4 + i] + b[4 + i]);
#pragma unroll
        for (int i = 0; i < 16; ++i) res[i] = __expf(rinv * a2 * s_out[8 + i] + b[8 + i]);
        for (int it = 0; it < 16; ++it) {        // SK_IT*SK_IT
#pragma unroll
            for (int i = 0; i < 4; ++i) { float s = 0; for (int j = 0; j < 4; ++j) s += res[i*4+j] * v[j]; u[i] = 1.0f / (s + 1e-12f); }
#pragma unroll
            for (int j = 0; j < 4; ++j) { float s = 0; for (int i = 0; i < 4; ++i) s += res[i*4+j] * u[i]; v[j] = 1.0f / (s + 1e-12f); }
        }
#pragma unroll
        for (int i = 0; i < 16; ++i) hres[(long)t * 16 + i] = u[i >> 2] * res[i] * v[i & 3];
    }
    __syncthreads();

    float hv[4]; float ssq2 = 0.f;
#pragma unroll
    for (int q = 0; q < 4; ++q) {
        int c = tid + q * 256;
        float s = 0.f;
#pragma unroll
        for (int i = 0; i < 4; ++i) s += s_pre[i] * x[i * H_ + c];
        hv[q] = s; ssq2 += s * s;
    }
    red[tid] = ssq2; __syncthreads();
    for (int s = 128; s > 0; s >>= 1) { if (tid < s) red[tid] += red[tid + s]; __syncthreads(); }
    float rinv2 = rsqrtf(red[0] * (1.0f / H_) + 1e-6f);
#pragma unroll
    for (int q = 0; q < 4; ++q) {
        int c = tid + q * 256;
        hp_bf[(long)t * H_ + c] = f2bf(hv[q] * rinv2 * preg[c]);
    }
}

// ---------------------------------------------------------------------------
// Row RMS-norm (H=1024) -> bf16 (used for post-attn LN)
// ---------------------------------------------------------------------------
__global__ void rms_rows(const float* __restrict__ X, const float* __restrict__ g,
                         bf16_t* __restrict__ Y) {
    int t = blockIdx.x, tid = threadIdx.x;   // 256
    __shared__ float red[256];
    const float* x = X + (long)t * H_;
    float xv[4]; float s = 0.f;
#pragma unroll
    for (int q = 0; q < 4; ++q) { xv[q] = x[tid + q * 256]; s += xv[q] * xv[q]; }
    red[tid] = s; __syncthreads();
    for (int st = 128; st > 0; st >>= 1) { if (tid < st) red[tid] += red[tid + st]; __syncthreads(); }
    float rinv = rsqrtf(red[0] * (1.0f / H_) + 1e-6f);
#pragma unroll
    for (int q = 0; q < 4; ++q) { int c = tid + q * 256; Y[(long)t * H_ + c] = f2bf(xv[q] * rinv * g[c]); }
}

// ---------------------------------------------------------------------------
// SwiGLU combine: gu = bf16(silu(gate) * up)
// ---------------------------------------------------------------------------
__global__ void silu_mul(const float* __restrict__ gte, const float* __restrict__ up,
                         bf16_t* __restrict__ out, long n) {
    long i = (long)blockIdx.x * 256 + threadIdx.x;
    if (i >= n) return;
    out[i] = f2bf(siluf_(gte[i]) * up[i]);
}

// ---------------------------------------------------------------------------
// out[t,i,:] = sum_j h_res[t,i,j]*h[t,j,:] + h_post[t,i]*mlp[t,:]
// ---------------------------------------------------------------------------
__global__ void final_out(const float* __restrict__ h, const float* __restrict__ hres,
                          const float* __restrict__ hpost, const float* __restrict__ mlp,
                          float* __restrict__ out) {
    long idx = (long)blockIdx.x * 256 + threadIdx.x;
    if (idx >= (long)T_ * HC_ * H_) return;
    int  c = idx & (H_ - 1);
    long tmp = idx >> 10;
    int  i = tmp & (HC_ - 1);
    long t = tmp >> 2;
    const float* hr = hres + t * 16 + (long)i * 4;
    const float* ht = h + t * (HC_ * H_);
    float s = hr[0] * ht[c] + hr[1] * ht[H_ + c] + hr[2] * ht[2 * H_ + c] + hr[3] * ht[3 * H_ + c];
    out[idx] = s + hpost[t * HC_ + i] * mlp[t * H_ + c];
}

// ===========================================================================
extern "C" void kernel_launch(void* const* d_in, const int* in_sizes, int n_in,
                              void* d_out, int out_size, void* d_ws, size_t ws_size,
                              hipStream_t stream) {
    (void)in_sizes; (void)n_in; (void)out_size; (void)ws_size;
    const float* hid   = (const float*)d_in[0];
    const int*   ids   = (const int*)  d_in[1];
    const float* table = (const float*)d_in[2];
    const float* keyp  = (const float*)d_in[3];
    const float* kng   = (const float*)d_in[4];
    const float* qng   = (const float*)d_in[5];
    const float* vpw   = (const float*)d_in[6];
    const float* cw    = (const float*)d_in[7];
    const float* cng   = (const float*)d_in[8];
    const float* mhcw  = (const float*)d_in[9];
    const float* mhcb  = (const float*)d_in[10];
    const float* mhal  = (const float*)d_in[11];
    const float* preg  = (const float*)d_in[12];
    const float* postg = (const float*)d_in[13];
    const float* attw  = (const float*)d_in[14];
    const float* gw    = (const float*)d_in[15];
    const float* uw    = (const float*)d_in[16];
    const float* dw    = (const float*)d_in[17];
    float* out = (float*)d_out;

    unsigned char* ws = (unsigned char*)d_ws;
    size_t off = 0;
    auto take = [&](size_t bytes) -> size_t {
        size_t r = off; off += (bytes + 255) & ~(size_t)255; return r;
    };
    const size_t o_emb  = take((size_t)T_ * EENG_ * 2);          // emb bf16
    const size_t o_kpT  = take((size_t)HC_ * H_ * EENG_ * 2);    // key_projs^T bf16
    const size_t o_vpT  = take((size_t)H_ * EENG_ * 2);
    const size_t o_attT = take((size_t)H_ * H_ * 2);
    const size_t o_gT   = take((size_t)DFF_ * H_ * 2);
    const size_t o_uT   = take((size_t)DFF_ * H_ * 2);
    const size_t o_dT   = take((size_t)H_ * DFF_ * 2);
    const size_t o_keys = take((size_t)HC_ * T_ * H_ * 4);       // keys f32; later gate_o/up_o
    const size_t o_P    = take((size_t)T_ * H_ * 4);             // vproj out; later attn_o/mlp_o
    const size_t o_msq  = take((size_t)T_ * 4);
    const size_t o_gate = take((size_t)T_ * HC_ * 4);
    const size_t o_a    = take((size_t)T_ * HC_ * 4);
    const size_t o_hres = take((size_t)T_ * 16 * 4);
    const size_t o_hpst = take((size_t)T_ * HC_ * 4);
    const size_t o_h    = take((size_t)T_ * HC_ * H_ * 4);
    const size_t o_hp   = take((size_t)T_ * H_ * 2);
    const size_t o_hp2  = take((size_t)T_ * H_ * 2);
    const size_t o_gu   = take((size_t)T_ * DFF_ * 2);

    bf16_t* emb   = (bf16_t*)(ws + o_emb);
    bf16_t* kpT   = (bf16_t*)(ws + o_kpT);
    bf16_t* vpT   = (bf16_t*)(ws + o_vpT);
    bf16_t* attT  = (bf16_t*)(ws + o_attT);
    bf16_t* gT    = (bf16_t*)(ws + o_gT);
    bf16_t* uT    = (bf16_t*)(ws + o_uT);
    bf16_t* dT    = (bf16_t*)(ws + o_dT);
    float*  keys  = (float*)(ws + o_keys);
    float*  gateo = (float*)(ws + o_keys);                       // reuse after gate kernel
    float*  upo   = (float*)(ws + o_keys + (size_t)T_ * DFF_ * 4);
    float*  P     = (float*)(ws + o_P);
    float*  attno = (float*)(ws + o_P);                          // reuse after build_h
    float*  mlpo  = (float*)(ws + o_P);                          // reuse after rms_rows
    float*  msq   = (float*)(ws + o_msq);
    float*  gateb = (float*)(ws + o_gate);
    float*  ab    = (float*)(ws + o_a);
    float*  hres  = (float*)(ws + o_hres);
    float*  hpost = (float*)(ws + o_hpst);
    float*  hbuf  = (float*)(ws + o_h);
    bf16_t* hp_bf = (bf16_t*)(ws + o_hp);
    bf16_t* hp2_bf= (bf16_t*)(ws + o_hp2);
    bf16_t* gu_bf = (bf16_t*)(ws + o_gu);

    // --- weight prep (transpose + bf16) ---
    {
        int bl = (int)(((long)H_ * EENG_ + 255) / 256);
        for (int g = 0; g < HC_; ++g)
            transpose_bf16<<<bl, 256, 0, stream>>>(keyp + (size_t)g * EENG_ * H_,
                                                   kpT + (size_t)g * H_ * EENG_, EENG_, H_);
        transpose_bf16<<<bl, 256, 0, stream>>>(vpw, vpT, EENG_, H_);
        transpose_bf16<<<(int)(((long)H_ * H_ + 255) / 256), 256, 0, stream>>>(attw, attT, H_, H_);
        transpose_bf16<<<(int)(((long)DFF_ * H_ + 255) / 256), 256, 0, stream>>>(gw, gT, H_, DFF_);
        transpose_bf16<<<(int)(((long)DFF_ * H_ + 255) / 256), 256, 0, stream>>>(uw, uT, H_, DFF_);
        transpose_bf16<<<(int)(((long)H_ * DFF_ + 255) / 256), 256, 0, stream>>>(dw, dT, DFF_, H_);
    }

    // --- engram ---
    gather_emb<<<T_, 256, 0, stream>>>(ids, table, emb);
    // keys[g] = emb @ key_projs[g]   (batched over HC)
    gemm_bf16_wmma<<<dim3((T_/16)*(H_/64)/8, HC_), 256, 0, stream>>>(
        emb, kpT, keys, T_, H_, EENG_, (long long)H_ * EENG_, (long long)T_ * H_);
    // P = emb @ vproj_w
    gemm_bf16_wmma<<<dim3((T_/16)*(H_/64)/8, 1), 256, 0, stream>>>(
        emb, vpT, P, T_, H_, EENG_, 0, 0);
    row_meansq<<<T_, 256, 0, stream>>>(P, msq);
    gate_kernel<<<dim3(T_, HC_), 128, 0, stream>>>(keys, hid, kng, qng, msq, gateb, ab);
    build_h<<<(int)(((long)T_ * HC_ * H_) / 256), 256, 0, stream>>>(hid, P, gateb, ab, cw, cng, hbuf);

    // --- manifold hyper-connections + pre-mix + pre-LN ---
    mhc_kernel<<<T_, 256, 0, stream>>>(hbuf, mhcw, mhcb, mhal, preg, hres, hpost, hp_bf);

    // --- attn stub + post-LN ---
    gemm_bf16_wmma<<<dim3((T_/16)*(H_/64)/8, 1), 256, 0, stream>>>(
        hp_bf, attT, attno, T_, H_, H_, 0, 0);
    rms_rows<<<T_, 256, 0, stream>>>(attno, postg, hp2_bf);

    // --- SwiGLU MLP ---
    gemm_bf16_wmma<<<dim3((T_/16)*(DFF_/64)/8, 1), 256, 0, stream>>>(
        hp2_bf, gT, gateo, T_, DFF_, H_, 0, 0);
    gemm_bf16_wmma<<<dim3((T_/16)*(DFF_/64)/8, 1), 256, 0, stream>>>(
        hp2_bf, uT, upo, T_, DFF_, H_, 0, 0);
    silu_mul<<<(int)(((long)T_ * DFF_) / 256), 256, 0, stream>>>(gateo, upo, gu_bf, (long)T_ * DFF_);
    gemm_bf16_wmma<<<dim3((T_/16)*(H_/64)/8, 1), 256, 0, stream>>>(
        gu_bf, dT, mlpo, T_, H_, DFF_, 0, 0);

    // --- recombination ---
    final_out<<<(int)(((long)T_ * HC_ * H_) / 256), 256, 0, stream>>>(hbuf, hres, hpost, mlpo, out);
}
